// LPNormKernel_31138512896692
// MI455X (gfx1250) — compile-verified, matrix operand order checked by reading, not run on hardware
//
#include <hip/hip_runtime.h>
#include <hip/hip_bf16.h>
#include <stdint.h>

typedef __attribute__((ext_vector_type(2)))  float    v2f;
typedef __attribute__((ext_vector_type(4)))  float    v4f;
typedef __attribute__((ext_vector_type(8)))  float    v8f;
typedef __attribute__((ext_vector_type(16))) _Float16 v16h;
typedef int v4i_g __attribute__((vector_size(16)));   // matches builtin's pointee type

#define GAS __attribute__((address_space(1)))
#define LAS __attribute__((address_space(3)))

static constexpr int C_IN        = 64;
static constexpr int C_OUT       = 128;
static constexpr int POS_PER_BLK = 16;
static constexpr int NPOS        = 4 * 56 * 56;   // 12544

#if __has_builtin(__builtin_amdgcn_global_load_async_to_lds_b128)
#define HAVE_ASYNC_LDS 1
#else
#define HAVE_ASYNC_LDS 0
#warning "CDNA5 probe: __builtin_amdgcn_global_load_async_to_lds_b128 NOT available -> sync LDS staging"
#endif

#if __has_builtin(__builtin_amdgcn_wmma_f32_16x16x4_f32)
#define HAVE_WMMA_F32X4 1
#else
#define HAVE_WMMA_F32X4 0
#warning "CDNA5 probe: __builtin_amdgcn_wmma_f32_16x16x4_f32 NOT available -> f16 WMMA epilogue"
#endif

__device__ __forceinline__ void stage_b128(const float* __restrict__ g,
                                           float* __restrict__ l) {
#if HAVE_ASYNC_LDS
  __builtin_amdgcn_global_load_async_to_lds_b128((GAS v4i_g*)g,
                                                 (LAS v4i_g*)l,
                                                 /*offset=*/0, /*cpol=*/0);
#else
  *(v4f*)l = *(const v4f*)g;
#endif
}

__launch_bounds__(256)
__global__ void lpnorm_l1_wmma_kernel(const float* __restrict__ x,
                                      const float* __restrict__ w,
                                      const float* __restrict__ b,
                                      float* __restrict__ out) {
  __shared__ float lds_w[C_IN * C_OUT];        // 32 KB  (c-major rows of 128)
  __shared__ float lds_x[POS_PER_BLK * C_IN];  //  4 KB  (pos-major rows of 64)
  __shared__ float lds_b[C_OUT];               // 512 B

  const int tid    = threadIdx.x;
  const int p_base = blockIdx.x * POS_PER_BLK;

  // ---- async stage: weights (2048 x b128), x tile (256 x b128), bias (32 x b128)
  #pragma unroll
  for (int i = 0; i < 8; ++i) {
    const int idx = tid + i * 256;                       // 0..2047
    stage_b128(w + idx * 4, lds_w + idx * 4);
  }
  stage_b128(x + (size_t)p_base * C_IN + tid * 4, lds_x + tid * 4);
  if (tid < 32) stage_b128(b + tid * 4, lds_b + tid * 4);

#if HAVE_ASYNC_LDS
#if __has_builtin(__builtin_amdgcn_s_wait_asynccnt)
  __builtin_amdgcn_s_wait_asynccnt(0);
#else
  asm volatile("s_wait_asynccnt 0" ::: "memory");
#endif
#endif
  __syncthreads();

  // ---- wave tile: 16 positions x 16 outputs, accumulators in WMMA C layout:
  //      lane 0-15: N=lane,    rows M = 0..7  in acc[0..7]
  //      lane16-31: N=lane-16, rows M = 8..15 in acc[0..7]
  const int wave   = tid >> 5;       // 0..7 -> output tile
  const int lane   = tid & 31;
  const int nidx   = lane & 15;
  const int halfq  = lane >> 4;
  const int o_base = wave * 16;
  const int row0   = halfq * 8;

  float acc[8] = {0.f, 0.f, 0.f, 0.f, 0.f, 0.f, 0.f, 0.f};

  for (int c4 = 0; c4 < C_IN; c4 += 4) {
    v4f xv[8];
    #pragma unroll
    for (int r = 0; r < 8; ++r)     // broadcast LDS reads (same addr across 16 lanes)
      xv[r] = *(const v4f*)&lds_x[(row0 + r) * C_IN + c4];
    #pragma unroll
    for (int k = 0; k < 4; ++k) {
      const float wv = lds_w[(c4 + k) * C_OUT + o_base + nidx];
      #pragma unroll
      for (int r = 0; r < 8; ++r)
        acc[r] += __builtin_fabsf(xv[r][k] - wv);   // v_sub + v_add |src|
    }
  }

  v8f cacc;
  #pragma unroll
  for (int r = 0; r < 8; ++r) cacc[r] = acc[r];

  const float bval = lds_b[o_base + nidx];

  // ---- bias broadcast-add via WMMA: C[m,n] += A[m,0]*B[0,n] = 1*b[n]
  // A 16x4 layout: lanes 0-15 hold K=0,1 (VGPR0,1); lanes 16-31 hold K=2,3.
  // B 4x16 layout: lanes 0-15 VGPR0 = row K=0 (N=lane); zero elsewhere.
#if HAVE_WMMA_F32X4
  v2f amat, bmat;
  amat.x = (halfq == 0) ? 1.0f : 0.0f;  amat.y = 0.0f;
  bmat.x = (halfq == 0) ? bval : 0.0f;  bmat.y = 0.0f;
  cacc = __builtin_amdgcn_wmma_f32_16x16x4_f32(false, amat, false, bmat,
                                               (short)0, cacc, false, false);
#else
  v16h a16, b16;
  #pragma unroll
  for (int i = 0; i < 16; ++i) { a16[i] = (_Float16)0.f; b16[i] = (_Float16)0.f; }
  a16[0] = (halfq == 0) ? (_Float16)1.f  : (_Float16)0.f;   // A[m,0] = 1
  b16[0] = (halfq == 0) ? (_Float16)bval : (_Float16)0.f;   // B[0,n] = b[n]
  cacc = __builtin_amdgcn_wmma_f32_16x16x32_f16(false, a16, false, b16,
                                                (short)0, cacc, false, false);
#endif

  // ---- store: row m = row0+r, col = o_base+nidx (lanes 0-15 coalesce per row)
  float* op = out + (size_t)(p_base + row0) * C_OUT + o_base + nidx;
  #pragma unroll
  for (int r = 0; r < 8; ++r)
    op[(size_t)r * C_OUT] = cacc[r];
}

extern "C" void kernel_launch(void* const* d_in, const int* in_sizes, int n_in,
                              void* d_out, int out_size, void* d_ws, size_t ws_size,
                              hipStream_t stream) {
  const float* x = (const float*)d_in[0];   // (4,56,56,64) NHWC contiguous
  const float* w = (const float*)d_in[1];   // (64,128)
  const float* b = (const float*)d_in[2];   // (128,)
  float* out = (float*)d_out;               // (4,3136,128)

  dim3 grid(NPOS / POS_PER_BLK);            // 784 blocks
  dim3 block(256);                          // 8 wave32s
  hipLaunchKernelGGL(lpnorm_l1_wmma_kernel, grid, block, 0, stream, x, w, b, out);
}
